// Net_89026082111679
// MI455X (gfx1250) — compile-verified, matrix operand order checked by reading, not run on hardware
//
#include <hip/hip_runtime.h>
#include <cstdint>

// ---------------------------------------------------------------------------
// Bi-GCN forward for MI455X (gfx1250, wave32).
// BN stats -> sign binarize -> 3x normalized sparse aggregation -> bf16 WMMA
// classifier GEMM (A-fragment reuse across 3 N-tiles) -> log_softmax.
// ---------------------------------------------------------------------------

typedef __bf16 bf16_t;
typedef __attribute__((ext_vector_type(16))) bf16_t v16bf;
typedef __attribute__((ext_vector_type(8)))  float  v8f;

#define N_NODES   100000
#define N_FEAT    128
#define N_CLASSES 40
#define N_EDGES   1600000
#define K_HOPS    3
#define BN_EPS    1e-5f

#define M_TILES   (N_NODES / 16)   // 6250
#define N_TILES   3                // 40 classes padded to 48
#define N_PAD     48

__device__ __forceinline__ void atomAddF(float* p, float v) {
    // Lowers to hardware global_atomic_add_f32 (no CAS loop).
    (void)__hip_atomic_fetch_add(p, v, __ATOMIC_RELAXED, __HIP_MEMORY_SCOPE_AGENT);
}

// ---- init: zero BN accumulators, deg = 1.0 (self loops) --------------------
__global__ void k_init(float* __restrict__ dinv, float* __restrict__ sum,
                       float* __restrict__ sumsq) {
    int i = blockIdx.x * blockDim.x + threadIdx.x;
    if (i < N_NODES) dinv[i] = 1.0f;
    if (i < N_FEAT) { sum[i] = 0.0f; sumsq[i] = 0.0f; }
}

// ---- BatchNorm column statistics ------------------------------------------
// blockDim = 128 (one thread per feature, coalesced row reads).
__global__ void k_bn_stats(const float* __restrict__ x, float* __restrict__ sum,
                           float* __restrict__ sumsq) {
    int f = threadIdx.x;
    float a = 0.0f, a2 = 0.0f;
    for (int r = blockIdx.x; r < N_NODES; r += gridDim.x) {
        float v = x[(size_t)r * N_FEAT + f];
        a += v; a2 += v * v;
    }
    atomAddF(&sum[f], a);
    atomAddF(&sumsq[f], a2);
}

// ---- degree over dst (self-loop already counted via init) ------------------
__global__ void k_degree(const long long* __restrict__ ei, float* __restrict__ deg) {
    int e = blockIdx.x * blockDim.x + threadIdx.x;
    if (e >= N_EDGES) return;
    long long d = ei[(size_t)N_EDGES + e];  // edge_index[1][e]
    atomAddF(&deg[d], 1.0f);
}

__global__ void k_dinv(float* __restrict__ dinv) {
    int i = blockIdx.x * blockDim.x + threadIdx.x;
    if (i < N_NODES) dinv[i] = rsqrtf(dinv[i]);
}

// ---- binarize: sign((x - mu) * rsqrt(var + eps)) ---------------------------
__global__ void k_binact(const float* __restrict__ x, const float* __restrict__ sum,
                         const float* __restrict__ sumsq, float* __restrict__ h) {
    int i = blockIdx.x * blockDim.x + threadIdx.x;
    if (i >= N_NODES * N_FEAT) return;
    int f = i & (N_FEAT - 1);
    const float inv_n = 1.0f / (float)N_NODES;
    float mu  = sum[f] * inv_n;
    float var = sumsq[f] * inv_n - mu * mu;
    float xn  = (x[i] - mu) * rsqrtf(var + BN_EPS);
    h[i] = (xn > 0.0f) ? 1.0f : ((xn < 0.0f) ? -1.0f : 0.0f);
}

// ---- self-loop term: h_next = dinv[i]^2 * h_cur (float4) -------------------
__global__ void k_selfloop(const float* __restrict__ hc, const float* __restrict__ dinv,
                           float* __restrict__ hn) {
    int t = blockIdx.x * blockDim.x + threadIdx.x;  // N_NODES * 32 float4 chunks
    if (t >= N_NODES * 32) return;
    int node = t >> 5, c = t & 31;
    float w = dinv[node]; w *= w;
    float4 v = ((const float4*)(hc + (size_t)node * N_FEAT))[c];
    float4 o; o.x = w * v.x; o.y = w * v.y; o.z = w * v.z; o.w = w * v.w;
    ((float4*)(hn + (size_t)node * N_FEAT))[c] = o;
}

// ---- edge scatter: one wave per edge, lane = float4 feature chunk ----------
__global__ void k_edges(const long long* __restrict__ ei, const float* __restrict__ dinv,
                        const float* __restrict__ hc, float* __restrict__ hn) {
    unsigned t = blockIdx.x * blockDim.x + threadIdx.x;  // < E*32 = 51.2M, fits u32
    if (t >= (unsigned)N_EDGES * 32u) return;
    int e    = (int)(t >> 5);
    int lane = (int)(t & 31u);
    long long s = ei[e];
    long long d = ei[(size_t)N_EDGES + e];
    float w = dinv[s] * dinv[d];
    float4 v = ((const float4*)(hc + (size_t)s * N_FEAT))[lane];
    float* o = hn + (size_t)d * N_FEAT + (size_t)lane * 4;
    atomAddF(o + 0, w * v.x);
    atomAddF(o + 1, w * v.y);
    atomAddF(o + 2, w * v.z);
    atomAddF(o + 3, w * v.w);
}

// ---- f32 -> bf16 conversions ----------------------------------------------
__global__ void k_cvt_h(const float* __restrict__ h, bf16_t* __restrict__ hb) {
    int i = blockIdx.x * blockDim.x + threadIdx.x;
    if (i < N_NODES * N_FEAT) hb[i] = (bf16_t)h[i];
}

__global__ void k_cvt_W(const float* __restrict__ W, bf16_t* __restrict__ Wb) {
    int i = blockIdx.x * blockDim.x + threadIdx.x;  // N_PAD * N_FEAT
    if (i >= N_PAD * N_FEAT) return;
    int n = i >> 7;
    Wb[i] = (n < N_CLASSES) ? (bf16_t)W[i] : (bf16_t)0.0f;
}

// ---- classifier GEMM via v_wmma_f32_16x16x32_bf16 --------------------------
// logits[100000 x 40] = h_bf16[100000 x 128] @ W_bf16^T + b
// One wave per 16-row M-tile; the A fragment is reused across all three
// 16-column N-tiles (3 accumulators, 3 WMMAs per K-step, 12 WMMAs total).
__global__ void k_gemm_wmma(const bf16_t* __restrict__ hb, const bf16_t* __restrict__ Wb,
                            const float* __restrict__ bias, float* __restrict__ logits) {
    int lane  = threadIdx.x & 31;
    int mtile = blockIdx.x * (blockDim.x >> 5) + (threadIdx.x >> 5);
    if (mtile >= M_TILES) return;   // wave-uniform: EXEC stays all-ones for WMMA

    int m0  = mtile << 4;
    int hi  = lane >> 4;      // K-half select per ISA 16-bit A/B layout
    int l15 = lane & 15;

    const bf16_t* arow  = hb + (size_t)(m0 + l15) * N_FEAT;  // A row (M = l15)
    const bf16_t* brow0 = Wb + (size_t)(l15) * N_FEAT;       // B cols = W rows
    const bf16_t* brow1 = Wb + (size_t)(16 + l15) * N_FEAT;
    const bf16_t* brow2 = Wb + (size_t)(32 + l15) * N_FEAT;

    v8f acc0 = {0.f, 0.f, 0.f, 0.f, 0.f, 0.f, 0.f, 0.f};
    v8f acc1 = acc0, acc2 = acc0;
#pragma unroll
    for (int k0 = 0; k0 < N_FEAT; k0 += 32) {
        v16bf a, b0, b1, b2;
#pragma unroll
        for (int e = 0; e < 16; ++e) {
            // element e of the 16x32 bf16 fragment: K = (e&7) + 8*hi + 16*(e>>3)
            int kk = k0 + (e & 7) + (hi << 3) + ((e >> 3) << 4);
            a[e]  = arow[kk];
            b0[e] = brow0[kk];
            b1[e] = brow1[kk];
            b2[e] = brow2[kk];
        }
        acc0 = __builtin_amdgcn_wmma_f32_16x16x32_bf16(false, a, false, b0,
                                                       (short)0, acc0, false, false);
        acc1 = __builtin_amdgcn_wmma_f32_16x16x32_bf16(false, a, false, b1,
                                                       (short)0, acc1, false, false);
        acc2 = __builtin_amdgcn_wmma_f32_16x16x32_bf16(false, a, false, b2,
                                                       (short)0, acc2, false, false);
    }

#pragma unroll
    for (int ntile = 0; ntile < N_TILES; ++ntile) {
        v8f acc = (ntile == 0) ? acc0 : ((ntile == 1) ? acc1 : acc2);
        int cc = (ntile << 4) + l15;       // D: N = lane&15
        if (cc < N_CLASSES) {
            float bv = bias[cc];
#pragma unroll
            for (int r = 0; r < 8; ++r) {  // D: M = r + 8*(lane>>4)
                int rr = m0 + r + (hi << 3);
                logits[(size_t)rr * N_CLASSES + cc] = acc[r] + bv;
            }
        }
    }
}

// ---- log_softmax: one wave per row, 40 classes -----------------------------
__global__ void k_logsoftmax(const float* __restrict__ logits, float* __restrict__ out) {
    int lane = threadIdx.x & 31;
    int row  = blockIdx.x * (blockDim.x >> 5) + (threadIdx.x >> 5);
    if (row >= N_NODES) return;
    const float* lr = logits + (size_t)row * N_CLASSES;
    float v0 = lr[lane];                          // lane < 32 < 40, always valid
    bool  has2 = (lane + 32) < N_CLASSES;         // lanes 0..7
    float v1 = has2 ? lr[lane + 32] : -3.4e38f;
    float m = fmaxf(v0, v1);
#pragma unroll
    for (int o = 16; o > 0; o >>= 1) m = fmaxf(m, __shfl_xor(m, o, 32));
    float s = __expf(v0 - m) + (has2 ? __expf(v1 - m) : 0.0f);
#pragma unroll
    for (int o = 16; o > 0; o >>= 1) s += __shfl_xor(s, o, 32);
    float lse = m + __logf(s);
    float* orow = out + (size_t)row * N_CLASSES;
    orow[lane] = v0 - lse;
    if (has2) orow[lane + 32] = v1 - lse;
}

// ---------------------------------------------------------------------------
extern "C" void kernel_launch(void* const* d_in, const int* in_sizes, int n_in,
                              void* d_out, int out_size, void* d_ws, size_t ws_size,
                              hipStream_t stream) {
    const float*     x  = (const float*)d_in[0];
    const float*     W  = (const float*)d_in[1];
    const float*     b  = (const float*)d_in[2];
    const long long* ei = (const long long*)d_in[3];  // int64 [2][E]
    float* out = (float*)d_out;

    char* ws = (char*)d_ws;
    size_t off = 0;
    auto take = [&](size_t bytes) -> void* {
        void* p = ws + off;
        off = (off + bytes + 255) & ~(size_t)255;
        return p;
    };
    float*  sum    = (float*)take((size_t)N_FEAT * 4);
    float*  sumsq  = (float*)take((size_t)N_FEAT * 4);
    float*  dinv   = (float*)take((size_t)N_NODES * 4);
    float*  hA     = (float*)take((size_t)N_NODES * N_FEAT * 4);
    float*  hB     = (float*)take((size_t)N_NODES * N_FEAT * 4);
    bf16_t* hb     = (bf16_t*)take((size_t)N_NODES * N_FEAT * 2);
    bf16_t* Wb     = (bf16_t*)take((size_t)N_PAD * N_FEAT * 2);
    float*  logits = (float*)take((size_t)N_NODES * N_CLASSES * 4);

    k_init<<<(N_NODES + 255) / 256, 256, 0, stream>>>(dinv, sum, sumsq);
    k_bn_stats<<<512, 128, 0, stream>>>(x, sum, sumsq);
    k_degree<<<(N_EDGES + 255) / 256, 256, 0, stream>>>(ei, dinv);
    k_dinv<<<(N_NODES + 255) / 256, 256, 0, stream>>>(dinv);
    k_binact<<<(N_NODES * N_FEAT + 255) / 256, 256, 0, stream>>>(x, sum, sumsq, hA);

    float* cur = hA;
    float* nxt = hB;
    for (int hop = 0; hop < K_HOPS; ++hop) {
        k_selfloop<<<(N_NODES * 32 + 255) / 256, 256, 0, stream>>>(cur, dinv, nxt);
        unsigned tot = (unsigned)N_EDGES * 32u;
        k_edges<<<(tot + 255) / 256, 256, 0, stream>>>(ei, dinv, cur, nxt);
        float* t = cur; cur = nxt; nxt = t;
    }

    k_cvt_h<<<(N_NODES * N_FEAT + 255) / 256, 256, 0, stream>>>(cur, hb);
    k_cvt_W<<<(N_PAD * N_FEAT + 255) / 256, 256, 0, stream>>>(W, Wb);

    // One wave per M-tile (6250 waves), 4 waves per block.
    k_gemm_wmma<<<(M_TILES + 3) / 4, 128, 0, stream>>>(hb, Wb, b, logits);

    k_logsoftmax<<<(N_NODES + 3) / 4, 128, 0, stream>>>(logits, out);
}
